// EdgeAwareLinkPredictor_79096117723244
// MI455X (gfx1250) — matrix operand order repaired and checked by reading, not compiled, not run on hardware
//
#include <hip/hip_runtime.h>

typedef __attribute__((ext_vector_type(16))) __bf16 v16bf;
typedef __attribute__((ext_vector_type(8)))  float  v8f;

#define TPB          256
#define WAVES_PER_WG 8
#define KSTEPS       17            // 544 / 32
#define NTILES       8             // 128 / 16
#define W1_FRAG_ELEMS (KSTEPS * NTILES * 32 * 16)   // 69632 bf16 elems (=544*128)
#define SMEM_BYTES    (W1_FRAG_ELEMS * 2 + 128 * 4 + 128 * 4)  // 140288 B

// Load one A-fragment (16x32 bf16, this lane's 16 elements) from a gathered
// f32 row. CDNA5 16-bit A layout: lane l (row M = l&15) holds
// K = hi8 + {0..7} in elems 0..7 and K = 16 + hi8 + {0..7} in elems 8..15,
// hi8 = 8*(l>>4). Both runs are 32B-contiguous -> 2x2 global b128 loads.
__device__ __forceinline__ v16bf load_a_frag(const float* __restrict__ p,
                                             int colbase, int hi8) {
  const float4* q0 = (const float4*)(p + colbase + hi8);
  const float4* q1 = (const float4*)(p + colbase + 16 + hi8);
  float4 f0 = q0[0], f1 = q0[1];
  float4 f2 = q1[0], f3 = q1[1];
  v16bf a;
  a[0]  = (__bf16)f0.x; a[1]  = (__bf16)f0.y; a[2]  = (__bf16)f0.z; a[3]  = (__bf16)f0.w;
  a[4]  = (__bf16)f1.x; a[5]  = (__bf16)f1.y; a[6]  = (__bf16)f1.z; a[7]  = (__bf16)f1.w;
  a[8]  = (__bf16)f2.x; a[9]  = (__bf16)f2.y; a[10] = (__bf16)f2.z; a[11] = (__bf16)f2.w;
  a[12] = (__bf16)f3.x; a[13] = (__bf16)f3.y; a[14] = (__bf16)f3.z; a[15] = (__bf16)f3.w;
  return a;
}

// One K-step for a 32-edge block: each LDS B-fragment feeds TWO WMMAs.
__device__ __forceinline__ void do_kstep2(const unsigned short* __restrict__ w1f,
                                          int kk, int lane,
                                          const v16bf& a0, const v16bf& a1,
                                          v8f acc0[NTILES], v8f acc1[NTILES]) {
#pragma unroll
  for (int nt = 0; nt < NTILES; ++nt) {
    const v16bf* bp =
        (const v16bf*)(w1f + (((kk * NTILES + nt) * 32 + lane) << 4));
    v16bf b = *bp;  // 2x ds_load_b128, reused by both WMMAs below
    acc0[nt] = __builtin_amdgcn_wmma_f32_16x16x32_bf16(
        false, a0, false, b, (short)0, acc0[nt], false, false);
    acc1[nt] = __builtin_amdgcn_wmma_f32_16x16x32_bf16(
        false, a1, false, b, (short)0, acc1[nt], false, false);
  }
}

__global__ void __launch_bounds__(TPB, 1)
edge_mlp_wmma_kernel(const float* __restrict__ z,
                     const int*   __restrict__ eli,   // [2, E]
                     const float* __restrict__ ea,    // [E, 32]
                     const float* __restrict__ W1,    // [544, 128]
                     const float* __restrict__ b1,    // [128]
                     const float* __restrict__ W2,    // [128]
                     const float* __restrict__ b2,    // [1]
                     float* __restrict__ out, int E) {
  extern __shared__ unsigned char smem[];
  unsigned short* w1f = (unsigned short*)smem;              // bf16 B fragments
  float* b1s = (float*)(smem + W1_FRAG_ELEMS * 2);
  float* w2s = b1s + 128;

  // --- Build W1 B-fragments in LDS (bf16, fragment-ready swizzle) ---------
  // B 32x16 16-bit layout: lane l holds col N=l&15; elem e holds K=16*(e>>3)+8*(l>>4)+(e&7).
  for (int idx = threadIdx.x; idx < W1_FRAG_ELEMS; idx += TPB) {
    int e  = idx & 15;
    int ln = (idx >> 4) & 31;
    int fi = idx >> 9;       // kk*8 + nt
    int nt = fi & 7;
    int kk = fi >> 3;
    int N  = nt * 16 + (ln & 15);
    int K  = ((e >> 3) << 4) + ((ln >> 4) << 3) + (e & 7);
    int f  = kk * 32 + K;
    ((__bf16*)w1f)[idx] = (__bf16)W1[f * 128 + N];
  }
  if (threadIdx.x < 128) {
    b1s[threadIdx.x] = b1[threadIdx.x];
    w2s[threadIdx.x] = W2[threadIdx.x];
  }
  __syncthreads();

  const int lane = threadIdx.x & 31;
  const int wave = threadIdx.x >> 5;
  const int hi8  = (lane >> 4) << 3;  // 0 or 8
  const int m    = lane & 15;         // row (edge) within a 16-edge sub-block
  const float b2v = b2[0];
  const int NB = (E + 31) >> 5;       // 32-edge blocks per wave iteration

  for (int blk = blockIdx.x * WAVES_PER_WG + wave; blk < NB;
       blk += (int)gridDim.x * WAVES_PER_WG) {
    const int e0 = blk << 5;
    int ge0 = e0 + m;       if (ge0 >= E) ge0 = E - 1;
    int ge1 = e0 + 16 + m;  if (ge1 >= E) ge1 = E - 1;
    const long long s0 = eli[ge0], d0 = eli[E + ge0];
    const long long s1 = eli[ge1], d1 = eli[E + ge1];
    const float* ps0 = z + s0 * 256;
    const float* pd0 = z + d0 * 256;
    const float* pe0 = ea + (long long)ge0 * 32;
    const float* ps1 = z + s1 * 256;
    const float* pd1 = z + d1 * 256;
    const float* pe1 = ea + (long long)ge1 * 32;

    v8f acc0[NTILES] = {};
    v8f acc1[NTILES] = {};

#pragma unroll 1
    for (int kk = 0; kk < 8; ++kk) {        // feat[0:256)   = z[src]
      v16bf a0 = load_a_frag(ps0, kk * 32, hi8);
      v16bf a1 = load_a_frag(ps1, kk * 32, hi8);
      do_kstep2(w1f, kk, lane, a0, a1, acc0, acc1);
    }
#pragma unroll 1
    for (int kk = 8; kk < 16; ++kk) {       // feat[256:512) = z[dst]
      v16bf a0 = load_a_frag(pd0, (kk - 8) * 32, hi8);
      v16bf a1 = load_a_frag(pd1, (kk - 8) * 32, hi8);
      do_kstep2(w1f, kk, lane, a0, a1, acc0, acc1);
    }
    {                                       // feat[512:544) = edge_attr
      v16bf a0 = load_a_frag(pe0, 0, hi8);
      v16bf a1 = load_a_frag(pe1, 0, hi8);
      do_kstep2(w1f, 16, lane, a0, a1, acc0, acc1);
    }

    // --- Epilogue: relu(acc + b1) . W2, reduce over N ---------------------
    // C/D layout: lane l, reg v -> (M = v + 8*(l>>4), N = l&15).
    float sa[8], sb[8];
#pragma unroll
    for (int v = 0; v < 8; ++v) { sa[v] = 0.0f; sb[v] = 0.0f; }
#pragma unroll
    for (int nt = 0; nt < NTILES; ++nt) {
      int n = nt * 16 + m;
      float bias = b1s[n];
      float w2v  = w2s[n];
#pragma unroll
      for (int v = 0; v < 8; ++v) {
        float h0 = fmaxf(acc0[nt][v] + bias, 0.0f);
        float h1 = fmaxf(acc1[nt][v] + bias, 0.0f);
        sa[v] = fmaf(h0, w2v, sa[v]);
        sb[v] = fmaf(h1, w2v, sb[v]);
      }
    }
    // Sum the 16 partial N-sums within each half-wave.
#pragma unroll
    for (int v = 0; v < 8; ++v) {
      sa[v] += __shfl_xor(sa[v], 1, 32);
      sa[v] += __shfl_xor(sa[v], 2, 32);
      sa[v] += __shfl_xor(sa[v], 4, 32);
      sa[v] += __shfl_xor(sa[v], 8, 32);
      sb[v] += __shfl_xor(sb[v], 1, 32);
      sb[v] += __shfl_xor(sb[v], 2, 32);
      sb[v] += __shfl_xor(sb[v], 4, 32);
      sb[v] += __shfl_xor(sb[v], 8, 32);
    }
    if (m == 0) {  // lane 0 -> rows 0..7, lane 16 -> rows 8..15 of each half
      int base0 = e0 + hi8;
      int base1 = e0 + 16 + hi8;
#pragma unroll
      for (int v = 0; v < 8; ++v) {
        int ei0 = base0 + v;
        int ei1 = base1 + v;
        if (ei0 < E) out[ei0] = sa[v] + b2v;
        if (ei1 < E) out[ei1] = sb[v] + b2v;
      }
    }
  }
}

extern "C" void kernel_launch(void* const* d_in, const int* in_sizes, int n_in,
                              void* d_out, int out_size, void* d_ws,
                              size_t ws_size, hipStream_t stream) {
  const float* z   = (const float*)d_in[0];
  const int*   eli = (const int*)  d_in[1];
  const float* ea  = (const float*)d_in[2];
  const float* W1  = (const float*)d_in[3];
  const float* b1  = (const float*)d_in[4];
  const float* W2  = (const float*)d_in[5];
  const float* b2  = (const float*)d_in[6];
  float* out = (float*)d_out;

  int E = in_sizes[1] / 2;
  if (E <= 0) return;

  (void)d_ws; (void)ws_size; (void)n_in; (void)out_size;

  hipFuncSetAttribute((const void*)edge_mlp_wmma_kernel,
                      hipFuncAttributeMaxDynamicSharedMemorySize, SMEM_BYTES);

  int nb = (E + 31) >> 5;
  int grid = (nb + WAVES_PER_WG - 1) / WAVES_PER_WG;
  if (grid > 2048) grid = 2048;

  edge_mlp_wmma_kernel<<<grid, TPB, SMEM_BYTES, stream>>>(z, eli, ea, W1, b1,
                                                          W2, b2, out, E);
}
